// SCEncoder_22686017257975
// MI455X (gfx1250) — matrix operand order repaired and checked by reading, not compiled, not run on hardware
//
#include <hip/hip_runtime.h>

typedef unsigned int u32;
typedef unsigned short u16;
typedef __attribute__((ext_vector_type(8)))  float  v8f;
typedef __attribute__((ext_vector_type(16))) __bf16 v16bf;

static constexpr int D     = 128;
static constexpr int H     = 256;
static constexpr int NBLK  = 256;
static constexpr int KINFO = 128;
static constexpr int B     = 128;

#define DEV __device__ __forceinline__

DEV u16 f2bf(float f) {               // f32 -> bf16, round-to-nearest-even
  u32 u = __float_as_uint(f);
  u += 0x7fffu + ((u >> 16) & 1u);
  return (u16)(u >> 16);
}
DEV float bf2f(u16 h) { return __uint_as_float(((u32)h) << 16); }

union AV { v16bf v; u32 u[8]; };

DEV v8f wmma_bf16(const AV& a, const AV& b, v8f acc) {
  return __builtin_amdgcn_wmma_f32_16x16x32_bf16(false, a.v, false, b.v,
                                                 (short)0, acc, false, false);
}

// ---------------------------------------------------------------- prep ----

// e_buf[0][b,i,:] = E_obs[2,:]  (all u_bar == 2), stored bf16
__global__ void initE0K(const float* __restrict__ E_obs, u16* __restrict__ e0, int total) {
  int idx = blockIdx.x * blockDim.x + threadIdx.x;
  if (idx < total) e0[idx] = f2bf(E_obs[2 * D + (idx & (D - 1))]);
}

// pos2info[pos] = info index or -1; also bf16-convert the 2xD label table
__global__ void prepSmallK(const int* __restrict__ info_set,
                           const float* __restrict__ E_lab,
                           int* __restrict__ pos2info,
                           u16* __restrict__ E_lab_bf) {
  int t = threadIdx.x;              // 256 threads
  pos2info[t] = -1;
  E_lab_bf[t] = f2bf(E_lab[t]);     // 2*D == 256 elements
  __syncthreads();
  if (t < KINFO) pos2info[info_set[t]] = t;
}

// Swizzle W (Kin x Hout, row-major f32) into per-WMMA-tile bf16 B-operand layout:
// dst[((kt*NT + nt)*32 + lane)*8 + g] packs K = kt*32 + (lane>>4)*8 + (g>>2)*16 + 2(g&3) (+1 hi),
// N = nt*16 + (lane&15).  One contiguous 32B chunk per lane per tile (b128 loads).
__global__ void swizzleWK(const float* __restrict__ W, u32* __restrict__ dst,
                          int Hout, int total) {
  int NT = Hout >> 4;
  for (int idx = blockIdx.x * blockDim.x + threadIdx.x; idx < total;
       idx += gridDim.x * blockDim.x) {
    int g    = idx & 7;
    int lane = (idx >> 3) & 31;
    int t    = idx >> 8;
    int nt   = t % NT;
    int kt   = t / NT;
    int n = (nt << 4) + (lane & 15);
    int k = (kt << 5) + ((lane >> 4) << 3) + ((g >> 2) << 4) + ((g & 3) << 1);
    u16 lo = f2bf(W[k * Hout + n]);
    u16 hi = f2bf(W[(k + 1) * Hout + n]);
    dst[idx] = (u32)lo | ((u32)hi << 16);
  }
}

// ------------------------------------------------------------ fused MLP ----
// One wave (32 threads) per 16-row strip; used for all nodes with n >= 4.
// MODE 0 = checknode (KIN=2D), MODE 1 = bitnode (KIN=3D, uses u1h + E_lab_bf).
template <int KIN, int MODE>
__global__ void __launch_bounds__(32) mlpK(
    const u16* __restrict__ e_in, int n,
    const int* __restrict__ u1h,
    const u32* __restrict__ W1s, const float* __restrict__ b1,
    const u32* __restrict__ W2s, const float* __restrict__ b2,
    const u16* __restrict__ E_lab_bf,
    u16* __restrict__ out) {
  __shared__ u16 lds_a[16 * KIN];
  __shared__ u16 lds_h[16 * H];

  const int lane    = threadIdx.x;
  const int laneLo  = lane & 15;
  const int hiHalf  = lane >> 4;
  const int half    = n >> 1;
  const int rowBase = blockIdx.x << 4;

  // ---- stage 16 gathered input rows (bf16) into LDS (coalesced global reads)
  for (int idx = lane; idx < 16 * KIN; idx += 32) {
    int rr   = idx / KIN;
    int k    = idx - rr * KIN;
    int grow = rowBase + rr;
    int b    = grow / half;
    int i    = grow - b * half;
    u16 val;
    if (MODE == 0 || k < 2 * D) {
      int src = (k < D) ? ((b * n + 2 * i) * D + k)
                        : ((b * n + 2 * i + 1) * D + (k - D));
      val = e_in[src];
    } else {
      val = E_lab_bf[u1h[b * half + i] * D + (k - 2 * D)];
    }
    lds_a[idx] = val;
  }
  __syncthreads();

  // ---- layer 1: (16 x KIN) @ (KIN x H) + b1, ReLU -> lds_h (bf16)
  constexpr int KT1 = KIN / 32;
  #pragma unroll 1
  for (int nt = 0; nt < H / 16; ++nt) {
    float bias = b1[(nt << 4) + laneLo];
    v8f acc;
    #pragma unroll
    for (int v = 0; v < 8; ++v) acc[v] = bias;
    #pragma unroll
    for (int kt = 0; kt < KT1; ++kt) {
      AV a, bm;
      #pragma unroll
      for (int g = 0; g < 8; ++g) {        // A 16x32 bf16 lane layout (ISA 7.12.2)
        int kk = (kt << 5) + ((g & 3) << 1) + ((g >> 2) << 4) + (hiHalf << 3);
        a.u[g] = *(const u32*)&lds_a[laneLo * KIN + kk];
      }
      const u32* bp = W1s + ((((kt * (H / 16)) + nt) << 5) + lane) * 8;
      #pragma unroll
      for (int g = 0; g < 8; ++g) bm.u[g] = bp[g];
      acc = wmma_bf16(a, bm, acc);
    }
    int col = (nt << 4) + laneLo;          // D-matrix f32 layout: m = hiHalf*8+v
    #pragma unroll
    for (int v = 0; v < 8; ++v)
      lds_h[((hiHalf << 3) + v) * H + col] = f2bf(fmaxf(acc[v], 0.f));
  }
  __syncthreads();

  // ---- layer 2: (16 x H) @ (H x D) + b2 -> out (bf16)
  #pragma unroll 1
  for (int nt = 0; nt < D / 16; ++nt) {
    float bias = b2[(nt << 4) + laneLo];
    v8f acc;
    #pragma unroll
    for (int v = 0; v < 8; ++v) acc[v] = bias;
    #pragma unroll
    for (int kt = 0; kt < H / 32; ++kt) {
      AV a, bm;
      #pragma unroll
      for (int g = 0; g < 8; ++g) {
        int kk = (kt << 5) + ((g & 3) << 1) + ((g >> 2) << 4) + (hiHalf << 3);
        a.u[g] = *(const u32*)&lds_h[laneLo * H + kk];
      }
      const u32* bp = W2s + ((((kt * (D / 16)) + nt) << 5) + lane) * 8;
      #pragma unroll
      for (int g = 0; g < 8; ++g) bm.u[g] = bp[g];
      acc = wmma_bf16(a, bm, acc);
    }
    #pragma unroll
    for (int v = 0; v < 8; ++v)
      out[(rowBase + (hiHalf << 3) + v) * D + (nt << 4) + laneLo] = f2bf(acc[v]);
  }
}

// ---------------------------------------------- fused bottom (n==2) node ----
// Whole subtree in ONE kernel: checknode MLP -> left leaf -> bitnode MLP ->
// right leaf -> combine.  128 rows; 4 waves x 2 strips of 16 rows.
// A-operand fragments built directly from global bf16 as aligned u32 K-pairs
// (each 32-wide K-tile lies entirely inside one of the three D-sized sources).

template <int MODE>
DEV void bottom_strip(const u16* __restrict__ e_in,       // (B,2,D) bf16
                      const int* __restrict__ xl,         // LDS, B ints (MODE=1)
                      const u16* __restrict__ E_lab_bf,   // (2,D) bf16
                      const u32* __restrict__ W1s, const float* __restrict__ b1,
                      const u32* __restrict__ W2s, const float* __restrict__ b2,
                      u16* __restrict__ lds_h,            // per-wave 16*H
                      int rowBase, int lane,
                      u16* __restrict__ out) {            // (B,D) bf16
  constexpr int KIN = MODE ? 3 * D : 2 * D;
  constexpr int KT1 = KIN / 32;
  const int laneLo = lane & 15, hiHalf = lane >> 4;
  const int brow   = rowBase + laneLo;                    // half==1 -> b == row

  AV afrag[KT1];
  #pragma unroll
  for (int kt = 0; kt < KT1; ++kt) {
    #pragma unroll
    for (int g = 0; g < 8; ++g) {
      int k = (kt << 5) + ((g & 3) << 1) + ((g >> 2) << 4) + (hiHalf << 3);
      const u16* src;
      if (k < D)          src = &e_in[(brow * 2 + 0) * D + k];
      else if (k < 2 * D) src = &e_in[(brow * 2 + 1) * D + (k - D)];
      else                src = &E_lab_bf[xl[brow] * D + (k - 2 * D)];
      afrag[kt].u[g] = *(const u32*)src;
    }
  }
  // layer 1 + ReLU -> lds_h
  #pragma unroll 1
  for (int nt = 0; nt < H / 16; ++nt) {
    float bias = b1[(nt << 4) + laneLo];
    v8f acc;
    #pragma unroll
    for (int v = 0; v < 8; ++v) acc[v] = bias;
    #pragma unroll
    for (int kt = 0; kt < KT1; ++kt) {
      AV bm;
      const u32* bp = W1s + ((((kt * (H / 16)) + nt) << 5) + lane) * 8;
      #pragma unroll
      for (int g = 0; g < 8; ++g) bm.u[g] = bp[g];
      acc = wmma_bf16(afrag[kt], bm, acc);
    }
    int col = (nt << 4) + laneLo;
    #pragma unroll
    for (int v = 0; v < 8; ++v)
      lds_h[((hiHalf << 3) + v) * H + col] = f2bf(fmaxf(acc[v], 0.f));
  }
  __syncthreads();                        // uniform: every wave calls strip 2x
  // layer 2 -> out
  #pragma unroll 1
  for (int nt = 0; nt < D / 16; ++nt) {
    float bias = b2[(nt << 4) + laneLo];
    v8f acc;
    #pragma unroll
    for (int v = 0; v < 8; ++v) acc[v] = bias;
    #pragma unroll
    for (int kt = 0; kt < H / 32; ++kt) {
      AV a, bm;
      #pragma unroll
      for (int g = 0; g < 8; ++g) {
        int kk = (kt << 5) + ((g & 3) << 1) + ((g >> 2) << 4) + (hiHalf << 3);
        a.u[g] = *(const u32*)&lds_h[laneLo * H + kk];
      }
      const u32* bp = W2s + ((((kt * (D / 16)) + nt) << 5) + lane) * 8;
      #pragma unroll
      for (int g = 0; g < 8; ++g) bm.u[g] = bp[g];
      acc = wmma_bf16(a, bm, acc);
    }
    #pragma unroll
    for (int v = 0; v < 8; ++v)
      out[(rowBase + (hiHalf << 3) + v) * D + (nt << 4) + laneLo] = f2bf(acc[v]);
  }
}

DEV void leaf_eval(const u16* __restrict__ e_leaf, const float* __restrict__ Wl,
                   const float* __restrict__ bl, const float* __restrict__ r,
                   const int* __restrict__ info_bits,
                   const int* __restrict__ pos2info, int b, int pos,
                   int* x_out, float* p_out) {
  float z = bl[0];
  #pragma unroll 4
  for (int k = 0; k < D; ++k) z += bf2f(e_leaf[b * D + k]) * Wl[k];
  float p = 1.f / (1.f + __expf(-z));
  int hd  = (r[b * NBLK + pos] > p) ? 1 : 0;
  int fr  = fabsf(p - 0.5f) > 0.25f;
  int pi  = pos2info[pos];
  int fv  = (pi >= 0) ? info_bits[b * KINFO + pi] : 2;
  *x_out  = (fv == 2 || fr) ? hd : fv;
  *p_out  = p;
}

__global__ void __launch_bounds__(128) bottomK(
    const u16* __restrict__ e_in,      // ebuf[7]: (B,2,D) bf16
    u16* __restrict__ e_leaf,          // ebuf[8]: (B,D) bf16 scratch
    const u32* __restrict__ Wc1s, const float* __restrict__ bc1,
    const u32* __restrict__ Wc2s, const float* __restrict__ bc2,
    const u32* __restrict__ Wb1s, const float* __restrict__ bb1,
    const u32* __restrict__ Wb2s, const float* __restrict__ bb2,
    const u16* __restrict__ E_lab_bf,
    const float* __restrict__ Wl, const float* __restrict__ bl,
    const float* __restrict__ r, const int* __restrict__ info_bits,
    const int* __restrict__ pos2info, int lo,
    int* __restrict__ xdest,           // (B,2)
    float* __restrict__ out_u, float* __restrict__ out_p) {
  __shared__ u16 lds_h[4][16 * H];     // 32KB: per-wave hidden strip
  __shared__ int lds_xl[B];            // left-leaf decisions

  const int tid = threadIdx.x, wave = tid >> 5, lane = tid & 31;

  // phase 1: checknode MLP (128 rows) -> e_leaf
  for (int s = wave; s < 8; s += 4)
    bottom_strip<0>(e_in, nullptr, E_lab_bf, Wc1s, bc1, Wc2s, bc2,
                    lds_h[wave], s << 4, lane, e_leaf);
  __syncthreads();

  // phase 2: left leaf
  {
    int x; float p;
    leaf_eval(e_leaf, Wl, bl, r, info_bits, pos2info, tid, lo, &x, &p);
    lds_xl[tid] = x;
    out_u[tid * NBLK + lo] = (float)x;
    out_p[tid * NBLK + lo] = p;
  }
  __syncthreads();

  // phase 3: bitnode MLP -> e_leaf
  for (int s = wave; s < 8; s += 4)
    bottom_strip<1>(e_in, lds_xl, E_lab_bf, Wb1s, bb1, Wb2s, bb2,
                    lds_h[wave], s << 4, lane, e_leaf);
  __syncthreads();

  // phase 4: right leaf + combine
  {
    int x; float p;
    leaf_eval(e_leaf, Wl, bl, r, info_bits, pos2info, tid, lo + 1, &x, &p);
    out_u[tid * NBLK + lo + 1] = (float)x;
    out_p[tid * NBLK + lo + 1] = p;
    int xlv = lds_xl[tid];
    xdest[tid * 2 + 0] = (xlv ^ x) & 1;   // (u1h+u2h)%2
    xdest[tid * 2 + 1] = x;
  }
}

// ------------------------------------------------------------- combine ----
__global__ void combineK(const int* __restrict__ xl, const int* __restrict__ xr,
                         int half, int* __restrict__ xdst, int total) {
  int idx = blockIdx.x * blockDim.x + threadIdx.x;
  if (idx < total) {
    int b = idx / half, i = idx - b * half;
    int a = xl[idx], c = xr[idx];
    xdst[b * 2 * half + 2 * i]     = (a ^ c) & 1;
    xdst[b * 2 * half + 2 * i + 1] = c;
  }
}

// ------------------------------------------------------------ finalize ----
__global__ void finalizeK(const int* __restrict__ xroot,
                          const int* __restrict__ pos2info,
                          const float* __restrict__ r, float* __restrict__ out,
                          int total) {
  int idx = blockIdx.x * blockDim.x + threadIdx.x;
  if (idx < total) {
    out[idx]              = (float)xroot[idx];                       // x
    out[total + idx]      = (pos2info[idx % NBLK] >= 0) ? 2.f : 1.f; // f
    out[4 * total + idx]  = r[idx];                                  // r
    // u (seg 2) and p (seg 3) written by bottomK leaves
  }
}

// ---------------------------------------------------------------- host ----
struct Ctx {
  hipStream_t stream;
  const float *r, *bc1, *bc2, *bb1, *bb2, *Wl, *bl;
  const int* info_bits;
  const u32 *Wc1s, *Wb1s, *Wc2s, *Wb2s;
  const u16* E_lab_bf;
  const int* pos2info;
  u16* ebuf[9];
  int *xl[8], *xr[8];
  float *out_u, *out_p;
};

static void encode_node(const Ctx& c, int d, int lo, int n, int* xdest) {
  if (n == 2) {                                    // fused bottom subtree
    bottomK<<<1, 128, 0, c.stream>>>(
        c.ebuf[7], c.ebuf[8], c.Wc1s, c.bc1, c.Wc2s, c.bc2, c.Wb1s, c.bb1,
        c.Wb2s, c.bb2, c.E_lab_bf, c.Wl, c.bl, c.r, c.info_bits, c.pos2info,
        lo, xdest, c.out_u, c.out_p);
    return;
  }
  const int half = n >> 1;
  const int rows = B * half;
  mlpK<2 * D, 0><<<rows / 16, 32, 0, c.stream>>>(
      c.ebuf[d], n, nullptr, c.Wc1s, c.bc1, c.Wc2s, c.bc2, c.E_lab_bf,
      c.ebuf[d + 1]);
  encode_node(c, d + 1, lo, half, c.xl[d]);
  mlpK<3 * D, 1><<<rows / 16, 32, 0, c.stream>>>(
      c.ebuf[d], n, c.xl[d], c.Wb1s, c.bb1, c.Wb2s, c.bb2, c.E_lab_bf,
      c.ebuf[d + 1]);
  encode_node(c, d + 1, lo + half, half, c.xr[d]);
  combineK<<<(rows + 255) / 256, 256, 0, c.stream>>>(c.xl[d], c.xr[d], half,
                                                     xdest, rows);
}

extern "C" void kernel_launch(void* const* d_in, const int* in_sizes, int n_in,
                              void* d_out, int out_size, void* d_ws, size_t ws_size,
                              hipStream_t stream) {
  (void)in_sizes; (void)n_in; (void)out_size; (void)ws_size;
  const int*   info_bits = (const int*)d_in[0];
  const float* r         = (const float*)d_in[1];
  const int*   info_set  = (const int*)d_in[2];
  const float* E_obs     = (const float*)d_in[3];
  const float* E_lab     = (const float*)d_in[4];
  const float* Wc1       = (const float*)d_in[5];
  const float* bc1       = (const float*)d_in[6];
  const float* Wc2       = (const float*)d_in[7];
  const float* bc2       = (const float*)d_in[8];
  const float* Wb1       = (const float*)d_in[9];
  const float* bb1       = (const float*)d_in[10];
  const float* Wb2       = (const float*)d_in[11];
  const float* bb2       = (const float*)d_in[12];
  const float* Wl        = (const float*)d_in[13];
  const float* bl        = (const float*)d_in[14];
  float* out = (float*)d_out;

  char* p = (char*)d_ws;                        // deterministic bump allocator
  auto alloc = [&](size_t bytes) -> void* {
    void* ret = (void*)p;
    p += (bytes + 255) & ~(size_t)255;
    return ret;
  };

  Ctx c{};
  c.stream = stream;
  c.r = r; c.bc1 = bc1; c.bc2 = bc2; c.bb1 = bb1; c.bb2 = bb2;
  c.Wl = Wl; c.bl = bl; c.info_bits = info_bits;

  for (int d = 0; d <= 8; ++d)
    c.ebuf[d] = (u16*)alloc((size_t)B * (NBLK >> d) * D * sizeof(u16));
  for (int d = 0; d <= 7; ++d) {
    c.xl[d] = (int*)alloc((size_t)B * (128 >> d) * sizeof(int));
    c.xr[d] = (int*)alloc((size_t)B * (128 >> d) * sizeof(int));
  }
  int* xroot    = (int*)alloc((size_t)B * NBLK * sizeof(int));
  int* pos2info = (int*)alloc(NBLK * sizeof(int));
  u16* E_lab_bf = (u16*)alloc(2 * D * sizeof(u16));
  u32* Wc1s = (u32*)alloc((size_t)(2 * D / 32) * (H / 16) * 256 * sizeof(u32));
  u32* Wb1s = (u32*)alloc((size_t)(3 * D / 32) * (H / 16) * 256 * sizeof(u32));
  u32* Wc2s = (u32*)alloc((size_t)(H / 32) * (D / 16) * 256 * sizeof(u32));
  u32* Wb2s = (u32*)alloc((size_t)(H / 32) * (D / 16) * 256 * sizeof(u32));
  c.Wc1s = Wc1s; c.Wb1s = Wb1s; c.Wc2s = Wc2s; c.Wb2s = Wb2s;
  c.pos2info = pos2info;
  c.E_lab_bf = E_lab_bf;
  c.out_u = out + 2 * B * NBLK;
  c.out_p = out + 3 * B * NBLK;

  // ---- prep
  {
    int total = B * NBLK * D;
    initE0K<<<(total + 255) / 256, 256, 0, stream>>>(E_obs, c.ebuf[0], total);
  }
  prepSmallK<<<1, NBLK, 0, stream>>>(info_set, E_lab, pos2info, E_lab_bf);
  {
    int t1 = (2 * D / 32) * (H / 16) * 256;
    swizzleWK<<<(t1 + 255) / 256, 256, 0, stream>>>(Wc1, Wc1s, H, t1);
    int t2 = (3 * D / 32) * (H / 16) * 256;
    swizzleWK<<<(t2 + 255) / 256, 256, 0, stream>>>(Wb1, Wb1s, H, t2);
    int t3 = (H / 32) * (D / 16) * 256;
    swizzleWK<<<(t3 + 255) / 256, 256, 0, stream>>>(Wc2, Wc2s, D, t3);
    swizzleWK<<<(t3 + 255) / 256, 256, 0, stream>>>(Wb2, Wb2s, D, t3);
  }

  // ---- static DFS over the SC tree (structure is data-independent)
  encode_node(c, 0, 0, NBLK, xroot);

  // ---- finalize outputs x / f / r
  {
    int total = B * NBLK;
    finalizeK<<<(total + 255) / 256, 256, 0, stream>>>(xroot, pos2info, r, out,
                                                       total);
  }
}
